// OptimizedSparseAttention_27247272526237
// MI455X (gfx1250) — matrix operand order repaired and checked by reading, not compiled, not run on hardware
//
#include <hip/hip_runtime.h>
#include <hip/hip_bf16.h>

// Streaming block top-k sparse attention for MI455X (gfx1250).
// BH=32, N=4096, D=64, fp32 in/out. k_keep = 409, Q rows per WG = 16.
//
// Phase A: scores[16][4096] = (Q*scale) @ K^T via v_wmma_f32_16x16x32_bf16,
//          fp32 accumulate, rows kept in LDS (256 KB of the 320 KB WGP LDS).
// Phase B: per-row exact top-409 via radix-select on monotone uint mapping,
//          ballot-compaction of selected key indices (ascending, tie-exact),
//          softmax-weighted gather of V from L2.

#define BH_       32
#define NKEY      4096
#define DIM       64
#define QB        16          // q rows per workgroup
#define KKEEP     409         // min(max(1, int(4096*0.1)), 1024)
#define NWAVE     4
#define NTHREADS  (NWAVE * 32)
#define NTILES    (NKEY / 16)             // 256 key tiles of 16
#define TPW       (NTILES / NWAVE)        // 64 tiles per wave
#define LIST_STR  412                     // padded list stride (ints)

// LDS layout (dynamic):
#define OFF_SC    0                                   // float [QB][NKEY]   262144 B
#define OFF_QBF   (OFF_SC + QB * NKEY * 4)            // __bf16 [QB][DIM]     2048 B
#define OFF_LIST  (OFF_QBF + QB * DIM * 2)            // int [QB][LIST_STR]  26368 B
#define OFF_HIST  (OFF_LIST + QB * LIST_STR * 4)      // int [NWAVE][256]     4096 B
#define LDS_BYTES (OFF_HIST + NWAVE * 256 * 4)        // = 294656 B  (< 320 KB)

typedef __attribute__((ext_vector_type(16))) __bf16 v16bf;
typedef __attribute__((ext_vector_type(8)))  float  v8f;

static __device__ __forceinline__ int lane_id() {
  return __builtin_amdgcn_mbcnt_hi(~0u, __builtin_amdgcn_mbcnt_lo(~0u, 0u));
}
static __device__ __forceinline__ int shfl_i(int v, int src) {
  return __builtin_amdgcn_ds_bpermute(src << 2, v);   // wave32: idx bits [6:2]
}
static __device__ __forceinline__ float shfl_f(float v, int src) {
  return __int_as_float(__builtin_amdgcn_ds_bpermute(src << 2, __float_as_int(v)));
}
static __device__ __forceinline__ unsigned ballot32(bool p) {
  return __builtin_amdgcn_ballot_w32(p);
}
// Monotone (strictly increasing) float -> uint mapping for radix select.
static __device__ __forceinline__ unsigned fmap(float f) {
  unsigned b = __float_as_uint(f);
  return (b & 0x80000000u) ? ~b : (b | 0x80000000u);
}
static __device__ __forceinline__ void cvt4(v16bf& b, int base, float4 f) {
  b[base + 0] = (__bf16)f.x; b[base + 1] = (__bf16)f.y;
  b[base + 2] = (__bf16)f.z; b[base + 3] = (__bf16)f.w;
}

extern "C" __global__ __launch_bounds__(NTHREADS, 1)
void topk_attn_kernel(const float* __restrict__ q,
                      const float* __restrict__ k,
                      const float* __restrict__ v,
                      float* __restrict__ out) {
  extern __shared__ char lds_raw[];
  float*  sc   = (float*)(lds_raw + OFF_SC);    // [QB][NKEY] score rows
  __bf16* qbf  = (__bf16*)(lds_raw + OFF_QBF);  // [QB][DIM]  pre-scaled Q
  int*    list = (int*)(lds_raw + OFF_LIST);    // [QB][LIST_STR] kept key idx
  int*    hist = (int*)(lds_raw + OFF_HIST);    // [NWAVE][256]

  const int tid  = threadIdx.x;
  const int lane = lane_id();
  const int wave = tid >> 5;
  const int bh   = blockIdx.x / (NKEY / QB);
  const int qb   = blockIdx.x % (NKEY / QB);
  const int qr0  = qb * QB;

  const float* Qb = q + ((size_t)bh * NKEY + qr0) * DIM;
  const float* Kb = k + (size_t)bh * NKEY * DIM;
  const float* Vb = v + (size_t)bh * NKEY * DIM;

  // ---- Load Q tile, fold in 1/sqrt(D)=0.125 scale, convert to bf16 ----
  for (int i = tid; i < QB * DIM; i += NTHREADS)
    qbf[i] = (__bf16)(Qb[i] * 0.125f);
  __syncthreads();

  // ---- Phase A: scores via bf16 WMMA (16x16x32, fp32 accum) ----
  const int half = lane >> 4;   // ISA 16-bit A/B layout: lane halves
  const int sub  = lane & 15;

  // A-matrix (16x32 bf16): lane holds row M=sub; element e maps to
  // K = half*16 + e   (same k order as B)
  v16bf A0, A1;
  #pragma unroll
  for (int e = 0; e < 16; ++e) {
    const int kk = half * 16 + e;
    A0[e] = qbf[sub * DIM + kk];          // K-dim d = 0..31
    A1[e] = qbf[sub * DIM + 32 + kk];     // K-dim d = 32..63
  }

  for (int t = 0; t < TPW; ++t) {
    const int tile   = wave * TPW + t;
    const int keyrow = tile * 16 + sub;   // B column n = sub -> this key row
    const float4* kp = (const float4*)(Kb + (size_t)keyrow * DIM);
    if (tile + 1 < NTILES)                // emits global_prefetch_b8
      __builtin_prefetch(Kb + (size_t)(keyrow + 16) * DIM, 0, 1);

    // B-matrix (32x16 bf16): lane holds col n=sub, K = half*16 + e.
    // B0 covers d = half*16 .. +15 ; B1 covers d = 32 + half*16 .. +15.
    v16bf B0, B1;
    cvt4(B0, 0, kp[half * 4 + 0]); cvt4(B0, 4, kp[half * 4 + 1]);
    cvt4(B0, 8, kp[half * 4 + 2]); cvt4(B0, 12, kp[half * 4 + 3]);
    cvt4(B1, 0, kp[8 + half * 4 + 0]); cvt4(B1, 4, kp[8 + half * 4 + 1]);
    cvt4(B1, 8, kp[8 + half * 4 + 2]); cvt4(B1, 12, kp[8 + half * 4 + 3]);

    v8f c = {};
    c = __builtin_amdgcn_wmma_f32_16x16x32_bf16(false, A0, false, B0,
                                                (short)0, c, false, false);
    c = __builtin_amdgcn_wmma_f32_16x16x32_bf16(false, A1, false, B1,
                                                (short)0, c, false, false);
    // C/D layout: VGPR r holds row M = r + half*8, column N = sub.
    #pragma unroll
    for (int r = 0; r < 8; ++r)
      sc[(size_t)(r + half * 8) * NKEY + tile * 16 + sub] = c[r];
  }
  __syncthreads();

  // ---- Phase B: per-row exact top-KKEEP + softmax-weighted V gather ----
  const unsigned below = (1u << lane) - 1u;
  for (int rr = 0; rr < 4; ++rr) {
    const int r = wave * 4 + rr;
    const float* srow = sc + (size_t)r * NKEY;

    // Row max (global max is always inside the kept set).
    float m = -1e30f;
    for (int i = 0; i < NKEY / 32; ++i) m = fmaxf(m, srow[i * 32 + lane]);
    #pragma unroll
    for (int off = 16; off > 0; off >>= 1) m = fmaxf(m, shfl_f(m, lane ^ off));

    // Radix-select the KKEEP-th largest value (in u-space), 4x 8-bit digits.
    int* h = hist + wave * 256;
    unsigned prefix = 0;
    int remaining = KKEEP;
    for (int level = 3; level >= 0; --level) {
      const int shift = level * 8;
      for (int i = lane; i < 256; i += 32) h[i] = 0;
      for (int i = 0; i < NKEY / 32; ++i) {
        const unsigned u = fmap(srow[i * 32 + lane]);
        const bool match =
            (level == 3) || ((u >> (shift + 8)) == (prefix >> (shift + 8)));
        if (match) atomicAdd(&h[(u >> shift) & 255], 1);
      }
      int hl[8]; int s8 = 0;
      #pragma unroll
      for (int d = 0; d < 8; ++d) { hl[d] = h[lane * 8 + d]; s8 += hl[d]; }
      int asc = s8;                             // ascending inclusive lane scan
      #pragma unroll
      for (int off = 1; off < 32; off <<= 1) {
        const int p = shfl_i(asc, lane - off);
        if (lane >= off) asc += p;
      }
      const int total = shfl_i(asc, 31);
      const int above = total - asc;            // counts in lanes > this lane
      const bool holds = (above < remaining) && (above + s8 >= remaining);
      const unsigned hm = ballot32(holds);
      const int src = 31 - __builtin_clz(hm);
      int digit = 0, newrem = remaining;
      if (lane == src) {
        int c = above;
        #pragma unroll
        for (int d = 7; d >= 0; --d) {
          c += hl[d];
          if (c >= remaining) {
            digit = lane * 8 + d; newrem = remaining - (c - hl[d]); break;
          }
        }
      }
      digit     = shfl_i(digit, src);
      remaining = shfl_i(newrem, src);
      prefix |= ((unsigned)digit) << shift;
    }
    const unsigned tu = prefix;  // u-map of k-th largest; `remaining` = ties needed

    // Ballot-compact kept indices: all u > tu, then `remaining` ties (u == tu),
    // in ascending key order (matches lax.top_k lowest-index tie-break).
    int* lrow = list + r * LIST_STR;
    int cnt = 0;
    for (int i = 0; i < NKEY / 32; ++i) {
      const int j = i * 32 + lane;
      const bool sel = fmap(srow[j]) > tu;
      const unsigned msk = ballot32(sel);
      if (sel) lrow[cnt + __builtin_popcount(msk & below)] = j;
      cnt += __builtin_popcount(msk);
    }
    int tneed = remaining;
    for (int i = 0; i < NKEY / 32 && tneed > 0; ++i) {
      const int j = i * 32 + lane;
      const bool sel = fmap(srow[j]) == tu;
      const unsigned msk = ballot32(sel);
      const int p = __builtin_popcount(msk & below);
      if (sel && p < tneed) lrow[cnt + p] = j;
      int take = __builtin_popcount(msk);
      take = take < tneed ? take : tneed;
      cnt += take; tneed -= take;
    }

    // Gather V rows (float2 per lane -> global_load_b64, L2-resident),
    // softmax-weighted accumulate, normalize.
    float acc0 = 0.f, acc1 = 0.f, denom = 0.f;
    const float2* V2 = (const float2*)Vb;
    for (int e = 0; e < KKEEP; ++e) {
      const int j = lrow[e];                    // LDS broadcast read
      const float w = __expf(srow[j] - m);      // uniform across lanes
      denom += w;
      const float2 vv = V2[(size_t)j * (DIM / 2) + lane];
      acc0 += w * vv.x;
      acc1 += w * vv.y;
    }
    const float inv = 1.0f / denom;
    float* orow = out + ((size_t)bh * NKEY + qr0 + r) * DIM;
    ((float2*)orow)[lane] = make_float2(acc0 * inv, acc1 * inv);
  }
}

extern "C" void kernel_launch(void* const* d_in, const int* in_sizes, int n_in,
                              void* d_out, int out_size, void* d_ws, size_t ws_size,
                              hipStream_t stream) {
  (void)in_sizes; (void)n_in; (void)out_size; (void)d_ws; (void)ws_size;
  const float* q = (const float*)d_in[0];
  const float* k = (const float*)d_in[1];
  const float* v = (const float*)d_in[2];
  float* out = (float*)d_out;
  dim3 grid(BH_ * (NKEY / QB));   // 32 * 256 = 8192 workgroups
  dim3 block(NTHREADS);           // 128 threads = 4 waves (wave32)
  hipLaunchKernelGGL(topk_attn_kernel, grid, block, LDS_BYTES, stream,
                     q, k, v, out);
}
